// VariationalMADE_57775900066318
// MI455X (gfx1250) — matrix operand order repaired and checked by reading, not compiled, not run on hardware
//
#include <hip/hip_runtime.h>
#include <hip/hip_bf16.h>

#define D_LAT 256
#define H_HID 1024
#define N_SMP 16384
#define DM1   255

typedef __bf16 bf16_t;
typedef __attribute__((ext_vector_type(16))) __bf16 v16bf;
typedef __attribute__((ext_vector_type(8)))  __bf16 v8bf;
typedef __attribute__((ext_vector_type(8)))  float  v8f;
typedef unsigned int v4u __attribute__((ext_vector_type(4)));
typedef int          v8i __attribute__((ext_vector_type(8)));
typedef int          v4i __attribute__((ext_vector_type(4)));

union ABfrag { v16bf v; v8bf h[2]; };

// TDM pads 16B after every 1KB of payload (pad_interval=7 -> 256 DWORDs,
// pad_amount=3 -> 4 DWORDs). Map linear payload byte p -> padded LDS byte.
__device__ __forceinline__ unsigned ldsPadded(unsigned p) {
    return p + ((p >> 10) << 4);
}

// ---------------------------------------------------------------------------
// Register-blocked bf16 WMMA GEMM: out[m,n] = act(sum_k A[m,k]*W[n,k] + b[n])
//   A [M x K] bf16 row-major (activations), streamed via global_load_b128
//   W [Nout x K] bf16 row-major (masked weights, ldw == K)
//   B panel (NT*16 weight rows x K) DMA'd to LDS once per block by the TDM.
// Each wave owns a 64x(NT*16) C macro-tile = 4 x NT WMMA accumulators.
// Block = 8 waves -> 512 rows x NT*16 cols.
// K is compile-time; the K-loop runs in 512-element chunks so LDS pad math
// hoists out of the inner loop (ds_load addresses become base + immediate).
// ---------------------------------------------------------------------------
template <int NT, int K>
__global__ __launch_bounds__(256)
void made_gemm_wmma(const bf16_t* __restrict__ A, int lda,
                    const bf16_t* __restrict__ W, int ldw,
                    const float*  __restrict__ bias, int Nout,
                    bf16_t* __restrict__ outBf, int ldob,
                    float*  __restrict__ outF,  int ldof,
                    int nColBase, int doRelu)
{
    constexpr int MT = 4;
    constexpr int CH = (K < 512) ? K : 512;              // K-chunk (1KB payload)
    constexpr unsigned PANEL_RAW = (unsigned)NT * 16u * (unsigned)K * 2u;
    constexpr unsigned PANEL     = PANEL_RAW + ((PANEL_RAW >> 10) << 4);
    __shared__ __align__(16) char bpanel[PANEL];

    const int lane = threadIdx.x;            // 0..31 (wave32)
    const int wave = threadIdx.y;            // 0..7
    const int nl   = lane & 15;
    const int hi   = lane >> 4;

    const int wBase = nColBase + blockIdx.x * (NT * 16);     // first weight row
    const int mWave = (blockIdx.y * 8 + wave) * (MT * 16);   // first A row

    // ---- Tensor Data Mover: B panel -> LDS (one descriptor per block) ----
    if (wave == 0) {
        const unsigned ldsOff = (unsigned)(size_t)(&bpanel[0]);
        const unsigned long long ga =
            (unsigned long long)(size_t)(W + (size_t)wBase * ldw);
        const unsigned td0   = (unsigned)K;              // contiguous dim
        const unsigned td1   = (unsigned)(Nout - wBase); // remaining rows
        const unsigned tile0 = (unsigned)K;
        const unsigned tile1 = (unsigned)(NT * 16);
        const unsigned long long st0 = (unsigned long long)ldw;

        v4u g0;
        g0[0] = 1u;                                       // count=1 (user D#)
        g0[1] = ldsOff;                                   // lds_addr
        g0[2] = (unsigned)ga;                             // global_addr lo
        g0[3] = (unsigned)((ga >> 32) & 0x1FFFFFFu) | (2u << 30); // hi | type=2
        v8i g1;
        g1[0] = (int)((1u << 16)      // data_size = 2 bytes
                    | (1u << 20)      // pad_enable
                    | (7u << 22)      // pad_interval = 256 DWORDs
                    | (3u << 25));    // pad_amount  = 4 DWORDs (16B)
        g1[1] = (int)((td0 & 0xFFFFu) << 16);                      // dim0 lo16
        g1[2] = (int)((td0 >> 16) | ((td1 & 0xFFFFu) << 16));      // dim0hi|dim1lo
        g1[3] = (int)((td1 >> 16) | ((tile0 & 0xFFFFu) << 16));    // dim1hi|tile0
        g1[4] = (int)(tile1 & 0xFFFFu);                            // tile1, tile2=0
        g1[5] = (int)(unsigned)st0;                                // dim0 stride lo
        g1[6] = (int)(unsigned)((st0 >> 32) & 0xFFFFu);            // stride hi
        g1[7] = 0;
        v4i gz4 = {0, 0, 0, 0};
        v8i gz8 = {0, 0, 0, 0, 0, 0, 0, 0};
        __builtin_amdgcn_tensor_load_to_lds(g0, g1, gz4, gz4, gz8, 0);
        __builtin_amdgcn_s_wait_tensorcnt(0);
    }
    __syncthreads();

    // ---- fragment addressing (CDNA5 16-bit A/B WMMA layouts) ----
    const int aSel = hi * 8;     // A: lanes 0-15 K+0..7/16..23, 16-31 K+8..15/24..31
    const int bSel = hi * 8;     // B: lanes 0-15 K+0..7/16..23, 16-31 K+8..15/24..31

    const bf16_t* __restrict__ aRow[MT];
    #pragma unroll
    for (int mt = 0; mt < MT; ++mt)
        aRow[mt] = A + (size_t)(mWave + mt * 16 + nl) * lda;

    unsigned pB[NT];                         // per-lane payload base of B column
    #pragma unroll
    for (int nt = 0; nt < NT; ++nt)
        pB[nt] = (unsigned)(nt * 16 + nl) * (unsigned)(K * 2);

    v8f acc[MT][NT];
    #pragma unroll
    for (int mt = 0; mt < MT; ++mt)
        #pragma unroll
        for (int nt = 0; nt < NT; ++nt)
            acc[mt][nt] = (v8f){0.f, 0.f, 0.f, 0.f, 0.f, 0.f, 0.f, 0.f};

    #pragma unroll
    for (int kc = 0; kc < K; kc += CH) {
        // 1KB payload windows never split a fragment for our shapes, so the
        // pad term is constant inside a chunk: hoist it here.
        const char* bBase[NT];
        #pragma unroll
        for (int nt = 0; nt < NT; ++nt)
            bBase[nt] = bpanel + ldsPadded(pB[nt] + 2u * (unsigned)kc)
                               + 2 * bSel;

        #pragma unroll 4
        for (int kk = 0; kk < CH; kk += 32) {
            ABfrag a[MT], b[NT];
            #pragma unroll
            for (int mt = 0; mt < MT; ++mt) {
                a[mt].h[0] = *(const v8bf*)(aRow[mt] + kc + kk + aSel);
                a[mt].h[1] = *(const v8bf*)(aRow[mt] + kc + kk + aSel + 16);
            }
            #pragma unroll
            for (int nt = 0; nt < NT; ++nt) {
                b[nt].h[0] = *(const v8bf*)(bBase[nt] + 2 * kk);
                b[nt].h[1] = *(const v8bf*)(bBase[nt] + 2 * kk + 32);
            }
            #pragma unroll
            for (int mt = 0; mt < MT; ++mt)
                #pragma unroll
                for (int nt = 0; nt < NT; ++nt)
                    acc[mt][nt] = __builtin_amdgcn_wmma_f32_16x16x32_bf16(
                        false, a[mt].v, false, b[nt].v, (short)0,
                        acc[mt][nt], false, false);
        }
    }

    // ---- epilogue: bias (+ReLU), bf16 activations and/or f32 logits ----
    #pragma unroll
    for (int nt = 0; nt < NT; ++nt) {
        const int wCol  = wBase + nt * 16 + nl;
        const int oColF = blockIdx.x * (NT * 16) + nt * 16 + nl;
        const float bv  = bias[wCol];
        #pragma unroll
        for (int mt = 0; mt < MT; ++mt) {
            #pragma unroll
            for (int v = 0; v < 8; ++v) {
                float r = acc[mt][nt][v] + bv;
                if (doRelu) r = fmaxf(r, 0.f);
                const int m = mWave + mt * 16 + v + hi * 8;  // C/D VGPR layout
                if (outBf) outBf[(size_t)m * ldob + wCol]  = (bf16_t)r;
                if (outF)  outF [(size_t)m * ldof + oColF] = r;
            }
        }
    }
}

// Fold MADE mask into weights, convert to bf16.
__global__ void made_prep_mask(const float* __restrict__ w,
                               bf16_t* __restrict__ out,
                               long total, int cols, int mode)
{
    long i = (long)blockIdx.x * blockDim.x + threadIdx.x;
    if (i >= total) return;
    int r = (int)(i / cols), c = (int)(i % cols);
    bool keep;
    if (mode == 0)      keep = (r % DM1 + 1) >= (c + 1);
    else if (mode == 1) keep = (r % DM1 + 1) >= (c % DM1 + 1);
    else                keep = (r + 1)       >  (c % DM1 + 1);
    out[i] = (bf16_t)(keep ? w[i] : 0.0f);
}

__global__ void made_f32_to_bf16(const float* __restrict__ in,
                                 bf16_t* __restrict__ out, long n)
{
    long i = (long)blockIdx.x * blockDim.x + threadIdx.x;
    if (i < n) out[i] = (bf16_t)in[i];
}

__global__ void made_zero_init(float* __restrict__ sOut,
                               bf16_t* __restrict__ sBf, long n)
{
    long i = (long)blockIdx.x * blockDim.x + threadIdx.x;
    if (i < n) { sOut[i] = 0.0f; sBf[i] = (bf16_t)0.0f; }
}

// One autoregressive step: sigmoid + deterministic-hash Bernoulli on column i.
__global__ void made_sample_step(const float* __restrict__ lt, int i,
                                 float* __restrict__ sOut,
                                 bf16_t* __restrict__ sBf)
{
    int n = blockIdx.x * blockDim.x + threadIdx.x;
    if (n >= N_SMP) return;
    float l = lt[(size_t)n * 16 + (i & 15)];
    float p = 1.0f / (1.0f + __expf(-l));
    unsigned h = (unsigned)n * 2654435761u ^ ((unsigned)i * 0x9E3779B9u) ^ 0xA511E9B3u;
    h ^= h >> 16; h *= 0x7feb352du; h ^= h >> 15; h *= 0x846ca68bu; h ^= h >> 16;
    float u = (float)(h >> 8) * (1.0f / 16777216.0f);
    float bit = (u < p) ? 1.0f : 0.0f;
    sOut[(size_t)n * D_LAT + i] = bit;
    sBf [(size_t)n * D_LAT + i] = (bf16_t)bit;
}

// logp[n] = sum_d x*l - softplus(l), one wave per row, lane-tree reduction.
__global__ __launch_bounds__(256)
void made_logp(const float* __restrict__ x, const float* __restrict__ logits,
               float* __restrict__ out)
{
    int n    = blockIdx.x * 8 + threadIdx.y;
    int lane = threadIdx.x;
    float s = 0.0f;
    for (int d = lane; d < D_LAT; d += 32) {
        float l  = logits[(size_t)n * D_LAT + d];
        float t  = x     [(size_t)n * D_LAT + d];
        float sp = fmaxf(l, 0.0f) + __logf(1.0f + __expf(-fabsf(l)));
        s += t * l - sp;
    }
    #pragma unroll
    for (int off = 16; off > 0; off >>= 1) s += __shfl_xor(s, off, 32);
    if (lane == 0) out[n] = s;
}

extern "C" void kernel_launch(void* const* d_in, const int* in_sizes, int n_in,
                              void* d_out, int out_size, void* d_ws, size_t ws_size,
                              hipStream_t stream)
{
    const float* x  = (const float*)d_in[0];
    const float* W1 = (const float*)d_in[1];
    const float* b1 = (const float*)d_in[2];
    const float* W2 = (const float*)d_in[3];
    const float* b2 = (const float*)d_in[4];
    const float* W3 = (const float*)d_in[5];
    const float* b3 = (const float*)d_in[6];

    float* sample = (float*)d_out;                                // [N, D]
    float* logp   = (float*)d_out + (size_t)N_SMP * D_LAT;        // [N]

    char*  ws  = (char*)d_ws;
    size_t off = 0;
    bf16_t* W1m = (bf16_t*)(ws + off); off += (size_t)H_HID * D_LAT * 2;
    bf16_t* W2m = (bf16_t*)(ws + off); off += (size_t)H_HID * H_HID * 2;
    bf16_t* W3m = (bf16_t*)(ws + off); off += (size_t)D_LAT * H_HID * 2;
    bf16_t* xbf = (bf16_t*)(ws + off); off += (size_t)N_SMP * D_LAT * 2;
    bf16_t* sbf = (bf16_t*)(ws + off); off += (size_t)N_SMP * D_LAT * 2;
    bf16_t* h1  = (bf16_t*)(ws + off); off += (size_t)N_SMP * H_HID * 2;
    bf16_t* h2  = (bf16_t*)(ws + off); off += (size_t)N_SMP * H_HID * 2;
    float*  lt  = (float*)(ws + off);  off += (size_t)N_SMP * 16 * 4;
    float*  lg  = (float*)(ws + off);  off += (size_t)N_SMP * D_LAT * 4;
    (void)in_sizes; (void)n_in; (void)out_size; (void)ws_size;

    const int thr = 256;
    made_prep_mask<<<(H_HID * D_LAT + thr - 1) / thr, thr, 0, stream>>>(
        W1, W1m, (long)H_HID * D_LAT, D_LAT, 0);
    made_prep_mask<<<(H_HID * H_HID + thr - 1) / thr, thr, 0, stream>>>(
        W2, W2m, (long)H_HID * H_HID, H_HID, 1);
    made_prep_mask<<<(D_LAT * H_HID + thr - 1) / thr, thr, 0, stream>>>(
        W3, W3m, (long)D_LAT * H_HID, H_HID, 2);
    made_f32_to_bf16<<<(N_SMP * D_LAT + thr - 1) / thr, thr, 0, stream>>>(
        x, xbf, (long)N_SMP * D_LAT);
    made_zero_init<<<(N_SMP * D_LAT + thr - 1) / thr, thr, 0, stream>>>(
        sample, sbf, (long)N_SMP * D_LAT);

    const dim3 blk(32, 8);                            // 8 waves (wave32)
    const dim3 gH(H_HID / 32, N_SMP / 512);           // hidden GEMMs, NT=2
    const dim3 g1(1,          N_SMP / 512);           // sampling column tile, NT=1
    const dim3 gD(D_LAT / 32, N_SMP / 512);           // full logits GEMM, NT=2

    // Serial autoregressive sampling: 256 steps, full masked forward each.
    for (int i = 0; i < D_LAT; ++i) {
        made_gemm_wmma<2, D_LAT><<<gH, blk, 0, stream>>>(
            sbf, D_LAT, W1m, D_LAT, b1, H_HID, h1, H_HID, nullptr, 0, 0, 1);
        made_gemm_wmma<2, H_HID><<<gH, blk, 0, stream>>>(
            h1, H_HID, W2m, H_HID, b2, H_HID, h2, H_HID, nullptr, 0, 0, 1);
        // Only the 16-wide column tile containing i is needed this step.
        made_gemm_wmma<1, H_HID><<<g1, blk, 0, stream>>>(
            h2, H_HID, W3m, H_HID, b3, D_LAT, nullptr, 0, lt, 16,
            (i >> 4) * 16, 0);
        made_sample_step<<<N_SMP / thr, thr, 0, stream>>>(lt, i, sample, sbf);
    }

    // Dense log-prob pass on x.
    made_gemm_wmma<2, D_LAT><<<gH, blk, 0, stream>>>(
        xbf, D_LAT, W1m, D_LAT, b1, H_HID, h1, H_HID, nullptr, 0, 0, 1);
    made_gemm_wmma<2, H_HID><<<gH, blk, 0, stream>>>(
        h1, H_HID, W2m, H_HID, b2, H_HID, h2, H_HID, nullptr, 0, 0, 1);
    made_gemm_wmma<2, H_HID><<<gD, blk, 0, stream>>>(
        h2, H_HID, W3m, H_HID, b3, D_LAT, nullptr, 0, lg, D_LAT, 0, 0);
    made_logp<<<N_SMP / 8, blk, 0, stream>>>(x, lg, logp);
}
